// DeepFactorRNN_82669530513626
// MI455X (gfx1250) — matrix operand order repaired and checked by reading, not compiled, not run on hardware
//
#include <hip/hip_runtime.h>
#include <hip/hip_bf16.h>
#include <cstdint>

// ---------- types ----------
typedef __attribute__((ext_vector_type(16))) __bf16 v16bf;
typedef __attribute__((ext_vector_type(8)))  float  v8f;

union Frag {
    uint32_t u[8];
    v16bf    v;
};

// ---------- helpers ----------
__device__ __forceinline__ uint16_t f2bf(float f) {
    uint32_t u = __float_as_uint(f);
    uint32_t r = u + 0x7FFFu + ((u >> 16) & 1u);   // round-to-nearest-even
    return (uint16_t)(r >> 16);
}
__device__ __forceinline__ float bf2f(uint16_t h) {
    return __uint_as_float(((uint32_t)h) << 16);
}
__device__ __forceinline__ float sigf(float x) {
    return 1.0f / (1.0f + __expf(-x));
}
__device__ __forceinline__ float tanh_fast(float x) {
    x = fminf(fmaxf(x, -15.0f), 15.0f);            // avoid inf/inf
    float e = __expf(2.0f * x);
    return (e - 1.0f) / (e + 1.0f);
}

// ---------- pack A: row-major (f32 or bf16) -> WMMA A-fragment order ----------
// Fragment dword (lane, v): row m = lane&15,
// K = kb*32 + (v>>2)*16 + (lane>>4)*8 + (v&3)*2 ; dword packs {K, K+1}.
// Storage: dst[((rb*Kb + kb)*256) + lane*8 + v]  (32B contiguous per lane -> b128 loads)
__global__ __launch_bounds__(256) void pack_a_kernel(
    const void* __restrict__ src, int srcIsF32,
    uint32_t* __restrict__ dst, int Kb, int K, size_t total)
{
    size_t idx = (size_t)blockIdx.x * blockDim.x + threadIdx.x;
    if (idx >= total) return;
    int s = (int)(idx & 255);
    size_t f = idx >> 8;
    int kb = (int)(f % Kb);
    int rb = (int)(f / Kb);
    int lane = s >> 3, v = s & 7;
    int m = rb * 16 + (lane & 15);
    int k = kb * 32 + ((v >> 2) << 4) + ((lane >> 4) << 3) + ((v & 3) << 1);
    uint16_t lo, hi;
    if (srcIsF32) {
        const float* p = (const float*)src + (size_t)m * K + k;
        lo = f2bf(p[0]); hi = f2bf(p[1]);
    } else {
        const uint16_t* p = (const uint16_t*)src + (size_t)m * K + k;
        lo = p[0]; hi = p[1];
    }
    dst[idx] = (uint32_t)lo | ((uint32_t)hi << 16);
}

// ---------- pack B: W (rows=4H, cols=K, f32 row-major) -> WMMA B-fragment order ----------
// B is 32x16 (KxN) per fragment: n = lane&15, K = kb*32 + (v>>2)*16 + (lane>>4)*8 + (v&3)*2 ; VGPR v packs K-pair {K, K+1}.
// B[k][n] = W[j0+n][k], so dword (lane,v) = { W[j0+n][K], W[j0+n][K+1] }.
__global__ __launch_bounds__(256) void pack_b_kernel(
    const float* __restrict__ W,
    uint32_t* __restrict__ dst, int Kb, int K, size_t total)
{
    size_t idx = (size_t)blockIdx.x * blockDim.x + threadIdx.x;
    if (idx >= total) return;
    int s = (int)(idx & 255);
    size_t f = idx >> 8;
    int kb = (int)(f % Kb);
    int cb = (int)(f / Kb);
    int lane = s >> 3, v = s & 7;
    int k = kb * 32 + ((v >> 2) << 4) + ((lane >> 4) << 3) + ((v & 3) << 1);
    int j = cb * 16 + (lane & 15);
    uint16_t lo = f2bf(W[(size_t)j * K + k]);
    uint16_t hi = f2bf(W[(size_t)j * K + k + 1]);
    dst[idx] = (uint32_t)lo | ((uint32_t)hi << 16);
}

// ---------- fragment load: 2 A row-blocks + 3 gate B blocks at K-step offset ----------
__device__ __forceinline__ void load_set(
    Frag& a0, Frag& a1, Frag& bi, Frag& bg, Frag& bo,
    const uint32_t* __restrict__ aP0, const uint32_t* __restrict__ aP1,
    const uint32_t* __restrict__ pI,  const uint32_t* __restrict__ pG,
    const uint32_t* __restrict__ pO,  size_t off)
{
#pragma unroll
    for (int v = 0; v < 8; ++v) {
        a0.u[v] = aP0[off + v];
        a1.u[v] = aP1[off + v];
        bi.u[v] = pI[off + v];
        bg.u[v] = pG[off + v];
        bo.u[v] = pO[off + v];
    }
}

#define WMMA6(A0, A1, BI, BG, BO)                                                             \
    ci0 = __builtin_amdgcn_wmma_f32_16x16x32_bf16(false, A0.v, false, BI.v, (short)0, ci0, false, false); \
    cg0 = __builtin_amdgcn_wmma_f32_16x16x32_bf16(false, A0.v, false, BG.v, (short)0, cg0, false, false); \
    co0 = __builtin_amdgcn_wmma_f32_16x16x32_bf16(false, A0.v, false, BO.v, (short)0, co0, false, false); \
    ci1 = __builtin_amdgcn_wmma_f32_16x16x32_bf16(false, A1.v, false, BI.v, (short)0, ci1, false, false); \
    cg1 = __builtin_amdgcn_wmma_f32_16x16x32_bf16(false, A1.v, false, BG.v, (short)0, cg1, false, false); \
    co1 = __builtin_amdgcn_wmma_f32_16x16x32_bf16(false, A1.v, false, BO.v, (short)0, co1, false, false);

// ---------- fused LSTM-step GEMM ----------
// pre = A @ W.T (+b), h = sig(o)*tanh(sig(i)*tanh(g)); forget gate skipped (c0 == 0).
// One wave computes a 32x16 output tile (2 row-blocks share the 3 gate-B fragments),
// with register double-buffering over the K dimension for load/WMMA overlap.
__global__ __launch_bounds__(256) void lstm_gemm_kernel(
    const uint32_t* __restrict__ afrag,
    const uint32_t* __restrict__ bfrag,   // all 4H columns (i|f|g|o blocks)
    const float* __restrict__ bias,       // 4H
    uint16_t* __restrict__ Hout,          // N x H bf16 row-major
    int colBlocks,                        // H/16
    int Kb, int H, int applyRelu, int numTiles)
{
    int lane = threadIdx.x & 31;
    int wave = threadIdx.x >> 5;
    int tile = blockIdx.x * 8 + wave;     // wave-uniform guard keeps EXEC all-1s for WMMA
    if (tile >= numTiles) return;
    int rp = tile / colBlocks;            // row-pair index (32 rows)
    int cb = tile % colBlocks;

    size_t laneOff = (size_t)lane << 3;
    size_t kstride = ((size_t)Kb) << 8;
    const uint32_t* aP0 = afrag + (size_t)(2 * rp) * kstride + laneOff;
    const uint32_t* aP1 = aP0 + kstride;
    const uint32_t* pI  = bfrag + (size_t)cb                    * kstride + laneOff;
    const uint32_t* pG  = bfrag + (size_t)(cb + 2 * colBlocks)  * kstride + laneOff;
    const uint32_t* pO  = bfrag + (size_t)(cb + 3 * colBlocks)  * kstride + laneOff;

    v8f ci0 = {}, cg0 = {}, co0 = {}, ci1 = {}, cg1 = {}, co1 = {};

    Frag A0a, A1a, BIa, BGa, BOa;   // buffer a
    Frag A0b, A1b, BIb, BGb, BOb;   // buffer b

    load_set(A0a, A1a, BIa, BGa, BOa, aP0, aP1, pI, pG, pO, 0);
    for (int kb = 0; kb < Kb; kb += 2) {
        load_set(A0b, A1b, BIb, BGb, BOb, aP0, aP1, pI, pG, pO, ((size_t)(kb + 1)) << 8);
        WMMA6(A0a, A1a, BIa, BGa, BOa)
        if (kb + 2 < Kb) {
            load_set(A0a, A1a, BIa, BGa, BOa, aP0, aP1, pI, pG, pO, ((size_t)(kb + 2)) << 8);
        }
        WMMA6(A0b, A1b, BIb, BGb, BOb)
    }

    // C/D layout: VGPR r, lane L -> row m = r + (L>>4)*8, col n = L&15
    int n = lane & 15, half = lane >> 4;
    int col = cb * 16 + n;
    float bi = bias[col];
    float bg = bias[2 * H + col];
    float bo = bias[3 * H + col];
    int row0 = rp * 32 + half * 8;
#pragma unroll
    for (int r = 0; r < 8; ++r) {
        {   // row-block 0
            float iv = ci0[r] + bi, gv = cg0[r] + bg, ov = co0[r] + bo;
            float hv = sigf(ov) * tanh_fast(sigf(iv) * tanh_fast(gv));
            if (applyRelu) hv = fmaxf(hv, 0.0f);
            Hout[(size_t)(row0 + r) * H + col] = f2bf(hv);
        }
        {   // row-block 1
            float iv = ci1[r] + bi, gv = cg1[r] + bg, ov = co1[r] + bo;
            float hv = sigf(ov) * tanh_fast(sigf(iv) * tanh_fast(gv));
            if (applyRelu) hv = fmaxf(hv, 0.0f);
            Hout[(size_t)(row0 + 16 + r) * H + col] = f2bf(hv);
        }
    }
}

// ---------- collapsed affine head prep: wsum = aff_W.sum(0), bsum = aff_b.sum() ----------
__global__ void head_prep_kernel(const float* __restrict__ affW,
                                 const float* __restrict__ affb,
                                 float* __restrict__ wsum, float* __restrict__ bsum)
{
    int j = threadIdx.x;  // 512 threads
    float s = 0.0f;
    for (int r = 0; r < 10; ++r) s += affW[r * 512 + j];
    wsum[j] = s;
    if (j == 0) {
        float b = 0.0f;
        for (int r = 0; r < 10; ++r) b += affb[r];
        *bsum = b;
    }
}

// ---------- final heads: mus = hg.wsum + bsum ; sigmas = softplus(hn.nW + nb) + 1e-6 ----------
__global__ __launch_bounds__(256) void head_final_kernel(
    const uint32_t* __restrict__ Hg2, const uint32_t* __restrict__ Hn2,   // bf16-pair dwords
    const float* __restrict__ wsum, const float* __restrict__ bsum,
    const float* __restrict__ nW, const float* __restrict__ nb,
    float* __restrict__ out, int N)
{
    int n = blockIdx.x * blockDim.x + threadIdx.x;
    if (n >= N) return;

    const uint32_t* hg = Hg2 + (size_t)n * 256;   // 512 bf16 = 256 dwords
    float acc = 0.0f;
#pragma unroll 4
    for (int j = 0; j < 256; ++j) {
        uint32_t w = hg[j];
        acc += bf2f((uint16_t)(w & 0xFFFFu)) * wsum[2 * j];
        acc += bf2f((uint16_t)(w >> 16))     * wsum[2 * j + 1];
    }
    out[n] = acc + *bsum;

    const uint32_t* hn = Hn2 + (size_t)n * 128;   // 256 bf16 = 128 dwords
    float a2 = 0.0f;
#pragma unroll 4
    for (int j = 0; j < 128; ++j) {
        uint32_t w = hn[j];
        a2 += bf2f((uint16_t)(w & 0xFFFFu)) * nW[2 * j];
        a2 += bf2f((uint16_t)(w >> 16))     * nW[2 * j + 1];
    }
    float z = a2 + nb[0];
    float sp = fmaxf(z, 0.0f) + log1pf(expf(-fabsf(z)));  // stable softplus
    out[N + n] = sp + 1e-6f;
}

extern "C" void kernel_launch(void* const* d_in, const int* in_sizes, int n_in,
                              void* d_out, int out_size, void* d_ws, size_t ws_size,
                              hipStream_t stream) {
    (void)in_sizes; (void)n_in; (void)out_size; (void)ws_size;

    const float* X      = (const float*)d_in[0];
    const float* gW0    = (const float*)d_in[1];
    const float* gb0    = (const float*)d_in[2];
    const float* gW1    = (const float*)d_in[3];
    const float* gb1    = (const float*)d_in[4];
    const float* affW   = (const float*)d_in[5];
    const float* affb   = (const float*)d_in[6];
    const float* nW0    = (const float*)d_in[7];
    const float* nb0    = (const float*)d_in[8];
    const float* nW1    = (const float*)d_in[9];
    const float* nb1    = (const float*)d_in[10];
    const float* noiseW = (const float*)d_in[11];
    const float* noiseb = (const float*)d_in[12];

    const int N = 128 * 256;   // 32768 rows
    const int F = 128, GH = 512, NH = 256;

    char* ws = (char*)d_ws;
    size_t off = 0;
    auto alloc = [&](size_t bytes) -> void* {
        void* p = ws + off;
        off = (off + bytes + 255) & ~(size_t)255;
        return p;
    };

    uint32_t* Xfrag = (uint32_t*)alloc((size_t)N * F * 2);        // A-frags of X (bf16)
    uint32_t* Bg0   = (uint32_t*)alloc((size_t)4 * GH * F * 2);
    uint32_t* Bg1   = (uint32_t*)alloc((size_t)4 * GH * GH * 2);
    uint32_t* Bn0   = (uint32_t*)alloc((size_t)4 * NH * F * 2);
    uint32_t* Bn1   = (uint32_t*)alloc((size_t)4 * NH * NH * 2);
    uint16_t* Hg1   = (uint16_t*)alloc((size_t)N * GH * 2);       // layer1 out (row-major bf16)
    uint32_t* Hg1f  = (uint32_t*)alloc((size_t)N * GH * 2);       // layer1 out as A-frags
    uint16_t* Hg2   = (uint16_t*)alloc((size_t)N * GH * 2);       // relu'd layer2 out
    uint16_t* Hn2   = (uint16_t*)alloc((size_t)N * NH * 2);
    float*    wsum  = (float*)alloc(512 * sizeof(float));
    float*    bsum  = (float*)alloc(sizeof(float));
    // noise branch layer-1 buffers alias the (by then dead) global-branch ones
    uint16_t* Hn1   = Hg1;
    uint32_t* Hn1f  = Hg1f;

    // --- pack inputs/weights into WMMA fragment layouts (bf16) ---
    {   size_t t = (size_t)N * F / 2;
        pack_a_kernel<<<(unsigned)((t + 255) / 256), 256, 0, stream>>>(X, 1, Xfrag, F / 32, F, t); }
    {   size_t t = (size_t)4 * GH * F / 2;
        pack_b_kernel<<<(unsigned)((t + 255) / 256), 256, 0, stream>>>(gW0, Bg0, F / 32, F, t); }
    {   size_t t = (size_t)4 * GH * GH / 2;
        pack_b_kernel<<<(unsigned)((t + 255) / 256), 256, 0, stream>>>(gW1, Bg1, GH / 32, GH, t); }
    {   size_t t = (size_t)4 * NH * F / 2;
        pack_b_kernel<<<(unsigned)((t + 255) / 256), 256, 0, stream>>>(nW0, Bn0, F / 32, F, t); }
    {   size_t t = (size_t)4 * NH * NH / 2;
        pack_b_kernel<<<(unsigned)((t + 255) / 256), 256, 0, stream>>>(nW1, Bn1, NH / 32, NH, t); }

    // --- global-factor branch: 2 fused LSTM-step GEMMs ---
    {   int tiles = (N / 32) * (GH / 16);
        lstm_gemm_kernel<<<(tiles + 7) / 8, 256, 0, stream>>>(Xfrag, Bg0, gb0, Hg1, GH / 16, F / 32, GH, 0, tiles); }
    {   size_t t = (size_t)N * GH / 2;
        pack_a_kernel<<<(unsigned)((t + 255) / 256), 256, 0, stream>>>(Hg1, 0, Hg1f, GH / 32, GH, t); }
    {   int tiles = (N / 32) * (GH / 16);
        lstm_gemm_kernel<<<(tiles + 7) / 8, 256, 0, stream>>>(Hg1f, Bg1, gb1, Hg2, GH / 16, GH / 32, GH, 1, tiles); }

    // --- noise branch ---
    {   int tiles = (N / 32) * (NH / 16);
        lstm_gemm_kernel<<<(tiles + 7) / 8, 256, 0, stream>>>(Xfrag, Bn0, nb0, Hn1, NH / 16, F / 32, NH, 0, tiles); }
    {   size_t t = (size_t)N * NH / 2;
        pack_a_kernel<<<(unsigned)((t + 255) / 256), 256, 0, stream>>>(Hn1, 0, Hn1f, NH / 32, NH, t); }
    {   int tiles = (N / 32) * (NH / 16);
        lstm_gemm_kernel<<<(tiles + 7) / 8, 256, 0, stream>>>(Hn1f, Bn1, nb1, Hn2, NH / 16, NH / 32, NH, 1, tiles); }

    // --- heads ---
    head_prep_kernel<<<1, 512, 0, stream>>>(affW, affb, wsum, bsum);
    head_final_kernel<<<(N + 255) / 256, 256, 0, stream>>>((const uint32_t*)Hg2, (const uint32_t*)Hn2,
                                                           wsum, bsum, noiseW, noiseb,
                                                           (float*)d_out, N);
}
